// AvgModel_39599598469804
// MI455X (gfx1250) — matrix operand order repaired and checked by reading, not compiled, not run on hardware
//
#include <hip/hip_runtime.h>
#include <math.h>

// ---------------------------------------------------------------------------
// AvgResNet2 forward for MI455X (gfx1250, wave32, WMMA bf16 16x16x32)
//
// Sizes: B=4, N=8192, M=B*N=32768, D=128, NB=15 (30 sub-layers), out=120
// ---------------------------------------------------------------------------

#define M_TOK   32768
#define D_FEAT  128
#define N_NODE  8192
#define B_BAT   4
#define EPSBN   1e-5f

typedef __attribute__((ext_vector_type(16))) __bf16 v16bf;
typedef __attribute__((ext_vector_type(8)))  float  v8f;

__device__ __forceinline__ float eluf(float x) {
    return x > 0.f ? x : expm1f(x);
}
__device__ __forceinline__ unsigned int f2bf_u16(float x) {
    unsigned int u = __float_as_uint(x);
    return (u + 0x7FFFu + ((u >> 16) & 1u)) >> 16;          // RNE to bf16
}
__device__ __forceinline__ unsigned int pack_bf2(float lo, float hi) {
    return f2bf_u16(lo) | (f2bf_u16(hi) << 16);
}

union FragAB { unsigned int u[8]; v16bf v; };

// ---------------------------------------------------------------------------
// One-time: convert weights to bf16, transposed [o][k].
// Slot s in 0..29 -> rn_W[s][0:128][:] ; slot 30 -> W2 (128x120) zero-padded.
// ---------------------------------------------------------------------------
__global__ void k_convert(const float* __restrict__ rnW,
                          const float* __restrict__ W2,
                          __bf16* __restrict__ Wt) {
    int gid = blockIdx.x * 256 + threadIdx.x;
    if (gid >= 31 * 16384) return;
    int s   = gid >> 14;
    int rem = gid & 16383;
    int o   = rem >> 7;
    int k   = rem & 127;
    float v;
    if (s < 30) v = rnW[((size_t)s * 256 + k) * 128 + o];
    else        v = (o < 120) ? W2[(size_t)k * 120 + o] : 0.f;
    unsigned int b = f2bf_u16(v);
    __bf16 h;
    *(unsigned short*)&h = (unsigned short)b;
    Wt[gid] = h;
}

// ---------------------------------------------------------------------------
// Per-batch mask sums (denominator of global average).
// ---------------------------------------------------------------------------
__global__ void k_masksum(const float* __restrict__ mask, float* __restrict__ msum) {
    __shared__ float sh[256];
    int b = blockIdx.x, tid = threadIdx.x;
    float s = 0.f;
    for (int i = tid; i < N_NODE; i += 256) s += mask[(size_t)b * N_NODE + i];
    sh[tid] = s; __syncthreads();
    for (int st = 128; st > 0; st >>= 1) {
        if (tid < st) sh[tid] += sh[tid + st];
        __syncthreads();
    }
    if (tid == 0) msum[b] = sh[0];
}

// ---------------------------------------------------------------------------
// conv1: x = inputs(32768x6) @ W1(6x128) + b1
// ---------------------------------------------------------------------------
__global__ void k_conv1(const float* __restrict__ inp, const float* __restrict__ W1,
                        const float* __restrict__ b1, float* __restrict__ X) {
    int gid = blockIdx.x * 256 + threadIdx.x;          // 32768*128
    int r = gid >> 7, o = gid & 127;
    float acc = b1[o];
    const float* row = inp + (size_t)r * 6;
#pragma unroll
    for (int i = 0; i < 6; ++i) acc += row[i] * W1[i * 128 + o];
    X[gid] = acc;
}

// ---------------------------------------------------------------------------
// Reduce: per 256-row block (within one batch) compute per-feature
//   sum(elu), sum(elu^2), sum(elu*mask)  ->  P[blk][3][128]
// Deterministic (no atomics).
// ---------------------------------------------------------------------------
__global__ void k_reduce(const float* __restrict__ A, const float* __restrict__ mask,
                         float* __restrict__ P) {
    __shared__ float sh[3][128];
    int tid = threadIdx.x;
    int f = tid & 127, rh = tid >> 7;
    int r0 = blockIdx.x * 256 + rh;
    float s = 0.f, s2 = 0.f, sm = 0.f;
    for (int k = 0; k < 128; ++k) {
        int r = r0 + 2 * k;
        float v = eluf(A[(size_t)r * 128 + f]);
        s += v; s2 += v * v; sm += v * mask[r];
    }
    if (rh == 1) { sh[0][f] = s; sh[1][f] = s2; sh[2][f] = sm; }
    __syncthreads();
    if (rh == 0) {
        s += sh[0][f]; s2 += sh[1][f]; sm += sh[2][f];
        P[((size_t)blockIdx.x * 3 + 0) * 128 + f] = s;
        P[((size_t)blockIdx.x * 3 + 1) * 128 + f] = s2;
        P[((size_t)blockIdx.x * 3 + 2) * 128 + f] = sm;
    }
}

// ---------------------------------------------------------------------------
// Finalize (residual sub-layer): BN stats for first 128 features, analytic BN
// over the 4 broadcast averages for features 128..255, and fold the second
// weight half into cvec[b][o].
//  stats layout: [0..127]=scale1, [128..255]=shift1, [256..767]=cvec[4][128]
// ---------------------------------------------------------------------------
__global__ void k_finalize(const float* __restrict__ P, const float* __restrict__ msum,
                           const float* __restrict__ gamma, const float* __restrict__ beta,
                           const float* __restrict__ Wfull, const float* __restrict__ bvec,
                           float* __restrict__ stats) {
    __shared__ float gn[4][128];
    int tid = threadIdx.x;
    if (tid < 128) {
        int f = tid;
        float s = 0.f, s2 = 0.f;
        for (int bk = 0; bk < 128; ++bk) {
            s  += P[((size_t)bk * 3 + 0) * 128 + f];
            s2 += P[((size_t)bk * 3 + 1) * 128 + f];
        }
        float sb[4];
#pragma unroll
        for (int b = 0; b < 4; ++b) {
            float t = 0.f;
            for (int q = 0; q < 32; ++q)
                t += P[((size_t)(b * 32 + q) * 3 + 2) * 128 + f];
            sb[b] = t;
        }
        const float invM = 1.f / (float)M_TOK;
        float mean = s * invM;
        float var  = s2 * invM - mean * mean;
        float sc   = rsqrtf(var + EPSBN) * gamma[f];
        stats[f]       = sc;
        stats[128 + f] = beta[f] - mean * sc;

        float ga[4];
#pragma unroll
        for (int b = 0; b < 4; ++b) ga[b] = sb[b] / msum[b];
        float m2 = 0.25f * (ga[0] + ga[1] + ga[2] + ga[3]);
        float v2 = 0.f;
#pragma unroll
        for (int b = 0; b < 4; ++b) { float d = ga[b] - m2; v2 += d * d; }
        v2 *= 0.25f;
        float sc2 = rsqrtf(v2 + EPSBN) * gamma[128 + f];
        float sh2 = beta[128 + f] - m2 * sc2;
#pragma unroll
        for (int b = 0; b < 4; ++b) gn[b][f] = ga[b] * sc2 + sh2;
    }
    __syncthreads();
    for (int idx = tid; idx < 512; idx += 256) {
        int b = idx >> 7, o = idx & 127;
        float acc = bvec[o];
        for (int f = 0; f < 128; ++f)
            acc += gn[b][f] * Wfull[(size_t)(128 + f) * 128 + o];
        stats[256 + idx] = acc;
    }
}

// ---------------------------------------------------------------------------
// Finalize (final head): BN(128) with g2/be2; cvec = b2 (zero-padded to 128).
// ---------------------------------------------------------------------------
__global__ void k_finalize2(const float* __restrict__ P, const float* __restrict__ g2,
                            const float* __restrict__ be2, const float* __restrict__ b2,
                            float* __restrict__ stats) {
    int tid = threadIdx.x;
    if (tid < 128) {
        int f = tid;
        float s = 0.f, s2 = 0.f;
        for (int bk = 0; bk < 128; ++bk) {
            s  += P[((size_t)bk * 3 + 0) * 128 + f];
            s2 += P[((size_t)bk * 3 + 1) * 128 + f];
        }
        const float invM = 1.f / (float)M_TOK;
        float mean = s * invM;
        float var  = s2 * invM - mean * mean;
        float sc   = rsqrtf(var + EPSBN) * g2[f];
        stats[f]       = sc;
        stats[128 + f] = be2[f] - mean * sc;
    }
    for (int idx = tid; idx < 512; idx += 256) {
        int o = idx & 127;
        stats[256 + idx] = (o < 120) ? b2[o] : 0.f;
    }
}

// ---------------------------------------------------------------------------
// Fused elu -> BN-affine -> bf16 -> GEMM(128x128 per tile) via WMMA.
// MODE 0: y = BN(elu(A)) @ W + cvec                      (write 128 cols)
// MODE 1: y = ... + Res                                  (residual add)
// MODE 2: final head: y = ... + tile(inputs[:, -3:],40)  (write 120 cols)
// Block: 256 threads (8 waves), computes 64 rows x 128 cols.
// ---------------------------------------------------------------------------
template <int MODE>
__global__ __launch_bounds__(256) void k_gemm(const float* __restrict__ A,
                                              const __bf16* __restrict__ Wt,
                                              const float* __restrict__ stats,
                                              const float* __restrict__ Res,
                                              float* __restrict__ Out,
                                              const float* __restrict__ inp) {
    __shared__ unsigned int ldsA[64 * 65];     // 64 rows x 128 bf16 (65 u32/row, padded)
    __shared__ unsigned int ldsW[128 * 65];    // 128 cols(o) x 128 bf16 over k
    __shared__ float lsS[128], lsH[128], lsC[128];

    const int tid = threadIdx.x;
    const int rowStart = blockIdx.x * 64;
    const int batch = rowStart >> 13;          // 8192 rows per batch

    if (tid < 128) {
        lsS[tid] = stats[tid];
        lsH[tid] = stats[128 + tid];
        lsC[tid] = stats[256 + batch * 128 + tid];
    }
    // stage W tile: 2048 uint4 loads, padded rows of 65 u32
    const uint4* Wg = reinterpret_cast<const uint4*>(Wt);
#pragma unroll
    for (int it = 0; it < 8; ++it) {
        int q = tid + it * 256;                // uint4 index
        uint4 w = Wg[q];
        int wb = q * 4;                        // u32 index
        int o = wb >> 6, wo = wb & 63;
        ldsW[o * 65 + wo + 0] = w.x;
        ldsW[o * 65 + wo + 1] = w.y;
        ldsW[o * 65 + wo + 2] = w.z;
        ldsW[o * 65 + wo + 3] = w.w;
    }
    __syncthreads();                           // lsS/lsH ready before A staging

    // stage A tile: elu -> affine -> bf16
    const float4* Ag = reinterpret_cast<const float4*>(A + (size_t)rowStart * 128);
#pragma unroll
    for (int it = 0; it < 8; ++it) {
        int q = tid + it * 256;                // float4 index
        float4 v = Ag[q];
        int fb = q * 4;
        int r = fb >> 7, f = fb & 127;
        float e0 = eluf(v.x) * lsS[f + 0] + lsH[f + 0];
        float e1 = eluf(v.y) * lsS[f + 1] + lsH[f + 1];
        float e2 = eluf(v.z) * lsS[f + 2] + lsH[f + 2];
        float e3 = eluf(v.w) * lsS[f + 3] + lsH[f + 3];
        ldsA[r * 65 + (f >> 1) + 0] = pack_bf2(e0, e1);
        ldsA[r * 65 + (f >> 1) + 1] = pack_bf2(e2, e3);
    }
    __syncthreads();

    const int lane = tid & 31;
    const int wv   = tid >> 5;
    const int rt   = wv & 3;                   // row tile 0..3
    const int cb   = (wv >> 2) * 4;            // col tiles cb..cb+3
    const int li   = lane & 15;
    const int hf   = lane >> 4;

    v8f c[4] = {v8f{0.f}, v8f{0.f}, v8f{0.f}, v8f{0.f}};

#pragma unroll
    for (int ks = 0; ks < 4; ++ks) {           // K = 4 x 32
        FragAB a;
        {
            const unsigned int base = (unsigned)(rt * 16 + li) * 65 + ks * 16 + hf * 8;
#pragma unroll
            for (int i = 0; i < 4; ++i) a.u[i]     = ldsA[base + i];
#pragma unroll
            for (int i = 0; i < 4; ++i) a.u[4 + i] = ldsA[base + 4 + i];
        }
#pragma unroll
        for (int t = 0; t < 4; ++t) {
            FragAB b;
            const unsigned int baseB = (unsigned)((cb + t) * 16 + li) * 65 + ks * 16 + hf * 8;
#pragma unroll
            for (int i = 0; i < 8; ++i) b.u[i] = ldsW[baseB + i];
            c[t] = __builtin_amdgcn_wmma_f32_16x16x32_bf16(
                false, a.v, false, b.v, (short)0, c[t], false, false);
        }
    }

    // epilogue
#pragma unroll
    for (int t = 0; t < 4; ++t) {
        const int col = (cb + t) * 16 + li;
        const float cc = lsC[col];
#pragma unroll
        for (int i = 0; i < 8; ++i) {
            const int m = i + hf * 8;
            const int grow = rowStart + rt * 16 + m;
            float val = c[t][i] + cc;
            if (MODE == 1) val += Res[(size_t)grow * 128 + col];
            if (MODE == 2) {
                if (col < 120) {
                    int c3 = col - (col / 3) * 3;
                    val += inp[(size_t)grow * 6 + 3 + c3];
                    Out[(size_t)grow * 120 + col] = val;
                }
            } else {
                Out[(size_t)grow * 128 + col] = val;
            }
        }
    }
}

// ---------------------------------------------------------------------------
// Host launcher
// ---------------------------------------------------------------------------
extern "C" void kernel_launch(void* const* d_in, const int* in_sizes, int n_in,
                              void* d_out, int out_size, void* d_ws, size_t ws_size,
                              hipStream_t stream) {
    const float* mask     = (const float*)d_in[1];
    const float* inputs   = (const float*)d_in[2];
    const float* W1       = (const float*)d_in[3];
    const float* b1       = (const float*)d_in[4];
    const float* rn_gamma = (const float*)d_in[5];
    const float* rn_beta  = (const float*)d_in[6];
    const float* rn_W     = (const float*)d_in[7];
    const float* rn_b     = (const float*)d_in[8];
    const float* g2       = (const float*)d_in[9];
    const float* be2      = (const float*)d_in[10];
    const float* W2       = (const float*)d_in[11];
    const float* b2       = (const float*)d_in[12];

    float* wsf = (float*)d_ws;
    float* X     = wsf;                         // 32768*128
    float* T     = wsf + 4194304;               // 32768*128
    float* P     = wsf + 8388608;               // 128*3*128
    float* stats = wsf + 8437760;               // 768
    float* msum  = wsf + 8438528;               // 4
    __bf16* Wt   = (__bf16*)(wsf + 8438532);    // 31*128*128 bf16 (16B aligned)

    k_convert<<<1984, 256, 0, stream>>>(rn_W, W2, Wt);
    k_masksum<<<4, 256, 0, stream>>>(mask, msum);
    k_conv1<<<16384, 256, 0, stream>>>(inputs, W1, b1, X);

    for (int blk = 0; blk < 15; ++blk) {
        int s0 = blk * 2;
        // sub-layer 1: X -> T
        k_reduce<<<128, 256, 0, stream>>>(X, mask, P);
        k_finalize<<<1, 256, 0, stream>>>(P, msum,
                                          rn_gamma + (size_t)s0 * 256,
                                          rn_beta  + (size_t)s0 * 256,
                                          rn_W     + (size_t)s0 * 256 * 128,
                                          rn_b     + (size_t)s0 * 128, stats);
        k_gemm<0><<<512, 256, 0, stream>>>(X, Wt + (size_t)s0 * 16384, stats,
                                           nullptr, T, nullptr);
        // sub-layer 2: T -> X (+ residual X)
        int s1 = s0 + 1;
        k_reduce<<<128, 256, 0, stream>>>(T, mask, P);
        k_finalize<<<1, 256, 0, stream>>>(P, msum,
                                          rn_gamma + (size_t)s1 * 256,
                                          rn_beta  + (size_t)s1 * 256,
                                          rn_W     + (size_t)s1 * 256 * 128,
                                          rn_b     + (size_t)s1 * 128, stats);
        k_gemm<1><<<512, 256, 0, stream>>>(T, Wt + (size_t)s1 * 16384, stats,
                                           X, X, nullptr);
    }

    // final head: elu -> BN(128) -> W2(+b2) + tiled input tail
    k_reduce<<<128, 256, 0, stream>>>(X, mask, P);
    k_finalize2<<<1, 256, 0, stream>>>(P, g2, be2, b2, stats);
    k_gemm<2><<<512, 256, 0, stream>>>(X, Wt + (size_t)30 * 16384, stats,
                                       nullptr, (float*)d_out, inputs);
}
